// IPEXLmHeadLinearAllreduceWithPaddingInt4_58626303591031
// MI455X (gfx1250) — matrix operand order repaired and checked by reading, not compiled, not run on hardware
//
#include <hip/hip_runtime.h>

// CDNA5 / gfx1250 int4-dequant LM-head GEMM: out[b,n] = sum_k x[b,-1,k]*(q-z)*s + bias
// Memory-bound: 263MB packed weights @ 23.3 TB/s ~= 11.5us floor.
// Design: pre-pass converts the last-token activations to f16 once (d_ws);
// then BOTH the weight tile and the activation tile are fetched per chunk by
// the Tensor Data Mover straight into LDS (hardware row padding gives the
// bank-conflict-free QROW=65 / XROW=40 layouts), triple-buffered at pipeline
// depth 2 so ~16KB of HBM stream is always in flight while
// v_wmma_f32_16x16x32_f16 consumes the previous tiles.

typedef __attribute__((ext_vector_type(16))) _Float16     v16h;
typedef __attribute__((ext_vector_type(8)))  _Float16     v8h;
typedef __attribute__((ext_vector_type(8)))  float        v8f;
typedef __attribute__((ext_vector_type(4)))  unsigned int v4u;
typedef __attribute__((ext_vector_type(4)))  int          v4i;
typedef __attribute__((ext_vector_type(8)))  int          v8i;

#define BATCH   32
#define SEQ     64
#define KDIM    4096
#define NPAD    32064
#define NPACK   (NPAD / 2)        // 16032 packed int4-pairs (one byte per int32) per row
#define NDIM    32000
#define NTILE   128               // output columns per block (8 waves x 16)
#define KCHUNK  32                // K per WMMA step
#define NCHUNK  (KDIM / KCHUNK)   // 128
#define QROW    65                // LDS row stride (ints): 64 data + 1 pad DW (TDM pad)
#define XROW    40                // LDS row stride (halfs): 32 data + 8 pad halfs (TDM pad)
#define NBUF    3                 // pipeline depth 2 -> triple buffer
#define QBYTES  (KCHUNK * QROW * 4)
#define XBYTES  (BATCH * XROW * 2)

#define HAVE_TDM __has_builtin(__builtin_amdgcn_tensor_load_to_lds)

#if HAVE_TDM
// Generic 2-D TDM descriptor: tile [tile1 rows x tile0 elems], element size
// 2^dsz_log bytes, global row stride stride0 elems, LDS padding pad_amt+1
// DWORDs after every 2^(pad_int+1) DWORDs.
__device__ __forceinline__ void tdm_load_2d(const void* gsrc, unsigned lds_off,
                                            int dsz_log, unsigned td0, unsigned td1,
                                            unsigned tile0, unsigned tile1,
                                            unsigned stride0, int pad_int, int pad_amt)
{
    const unsigned long long ga = (unsigned long long)gsrc;
    v4u g0;
    g0.x = 1u;                                        // count=1 (valid user descriptor)
    g0.y = lds_off;                                   // lds_addr
    g0.z = (unsigned)(ga & 0xffffffffu);              // global_addr[31:0]
    g0.w = (unsigned)(ga >> 32) | (2u << 30);         // global_addr[56:32] | type=2
    v8i g1;
    g1[0] = (dsz_log << 16) | (1 << 20) | (pad_int << 22) | (pad_amt << 25);
    g1[1] = (int)((td0 & 0xffffu) << 16);             // tensor_dim0[15:0]
    g1[2] = (int)((td0 >> 16) | ((td1 & 0xffffu) << 16));
    g1[3] = (int)((td1 >> 16) | (tile0 << 16));       // tensor_dim1[31:16] | tile_dim0
    g1[4] = (int)tile1;                               // tile_dim1 (tile_dim2=0)
    g1[5] = (int)stride0;                             // tensor_dim0_stride[31:0]
    g1[6] = 0;
    g1[7] = 0;
    v4i z4 = {0, 0, 0, 0};                            // 2-D: groups 2/3 NULL
#if __has_include(<hip/amd_detail/amd_gfx1250_TDM.h>)
    v8i z8 = {0, 0, 0, 0, 0, 0, 0, 0};                // clang-23 / therock: 6-arg form
    __builtin_amdgcn_tensor_load_to_lds(g0, g1, z4, z4, z8, 0);
#else
    __builtin_amdgcn_tensor_load_to_lds(g0, g1, z4, z4, 0);   // ROCm 7.2: 5-arg form
#endif
}
#endif

// ---- pre-pass: x[:, -1, :] f32 -> f16 into workspace [BATCH][KDIM] ----
__global__ __launch_bounds__(256)
void xcast_f16(const float* __restrict__ x, _Float16* __restrict__ xw)
{
    const int i  = blockIdx.x * 256 + threadIdx.x;    // 32 * 1024 float4s
    const int m  = i >> 10;
    const int c4 = (i & 1023) << 2;
    const float4 v = *(const float4*)(x + ((size_t)m * SEQ + (SEQ - 1)) * KDIM + c4);
    union { _Float16 h[4]; unsigned long long u; } p;
    p.h[0] = (_Float16)v.x; p.h[1] = (_Float16)v.y;
    p.h[2] = (_Float16)v.z; p.h[3] = (_Float16)v.w;
    *(unsigned long long*)(xw + (size_t)m * KDIM + c4) = p.u;
}

__global__ __launch_bounds__(256)
void int4_lmhead_wmma(const _Float16* __restrict__ xw,
                      const int*      __restrict__ qweight,
                      const float*    __restrict__ scales,
                      const int*      __restrict__ qzeros,
                      const float*    __restrict__ bias,
                      float*          __restrict__ out)
{
    __shared__ __align__(16) int      qs[NBUF][KCHUNK * QROW];  // packed weight tiles
    __shared__ __align__(16) _Float16 xs[NBUF][BATCH * XROW];   // activation tiles (f16)
    __shared__ float sc[2][NTILE];                              // per-column scale (group parity)
    __shared__ float zscl[2][NTILE];                            // per-column zero*scale

    const int tid  = threadIdx.x;
    const int lane = tid & 31;
    const int wave = tid >> 5;
    const int hi   = lane >> 4;
    const int ln   = lane & 15;
    const int n0   = blockIdx.x * NTILE;

    // hoisted LDS byte offsets (loop uses simple scalar adds, no readfirstlane churn)
    const unsigned qsbase = (unsigned)(size_t)&qs[0][0];
    const unsigned xsbase = (unsigned)(size_t)&xs[0][0];

    v8f acc0 = {};   // M = 0..15
    v8f acc1 = {};   // M = 16..31

    // ---- stage chunk t into buffer buf (rotating index, no modulo) ----
    auto stage = [&](int t, int buf) {
        const int k0 = t * KCHUNK;
        const int* qbase = qweight + (size_t)k0 * NPACK + (n0 >> 1);
#if HAVE_TDM
        if (tid < 32) {                               // wave0: two TDM issues per chunk
            tdm_load_2d(qbase, qsbase + (unsigned)buf * QBYTES,
                        /*4B*/2, NPACK, KDIM, 64, KCHUNK, NPACK, /*64DW*/5, /*+1DW*/0);
            tdm_load_2d(xw + k0, xsbase + (unsigned)buf * XBYTES,
                        /*2B*/1, KDIM, BATCH, KCHUNK, BATCH, KDIM, /*16DW*/3, /*+4DW*/3);
        }
#else
        #pragma unroll
        for (int i = 0; i < 2; ++i) {                 // fallback: b128 + ds_store
            const int idx = tid + i * 256;
            const int row = idx >> 4;
            const int c4  = (idx & 15) << 2;
            const int4 qv = *(const int4*)(qbase + (size_t)row * NPACK + c4);
            int* dst = &qs[buf][row * QROW + c4];
            dst[0] = qv.x; dst[1] = qv.y; dst[2] = qv.z; dst[3] = qv.w;
        }
        {
            const int m  = tid >> 3;
            const int c4 = (tid & 7) << 2;
            *(unsigned long long*)&xs[buf][m * XROW + c4] =
                *(const unsigned long long*)(xw + (size_t)m * KDIM + k0 + c4);
        }
#endif
        // group params at each group boundary, double-buffered by group parity
        if ((t & 3) == 0 && tid < NTILE) {
            const int g = t >> 2;
            const int p = g & 1;
            const int n = n0 + tid;
            const float s  = scales[(size_t)g * NPAD + n];
            const int   zb = qzeros[(size_t)g * NPACK + (n >> 1)];
            sc[p][tid]   = s;
            zscl[p][tid] = s * (float)((zb >> ((tid & 1) * 4)) & 15);
        }
    };

    stage(0, 0);
    stage(1, 1);
    int sb = 2;                                       // next buffer to stage into
    int cb = 0;                                       // buffer to compute from
    for (int t = 0; t < NCHUNK; ++t) {
        if (t + 2 < NCHUNK) {
            stage(t + 2, sb);                         // keep ~16KB of HBM stream in flight
            sb = (sb == NBUF - 1) ? 0 : sb + 1;
        }
#if HAVE_TDM
        if (tid < 32) {                               // wave0 owns TENSORcnt (in-order TDM)
            if (t + 2 < NCHUNK)      __builtin_amdgcn_s_wait_tensorcnt(4);  // chunk t landed
            else if (t + 1 < NCHUNK) __builtin_amdgcn_s_wait_tensorcnt(2);
            else                     __builtin_amdgcn_s_wait_tensorcnt(0);
        }
#endif
        __syncthreads();                              // staged tile t visible to all waves

        const int p = (t >> 2) & 1;

        // ---- B fragment: dequantized 32x16 f16 (k = hi*16 + j, same contiguous order as A) ----
        const int   nl = wave * 16 + ln;
        const int   c  = nl >> 1;
        const int   sh = (nl & 1) << 2;
        const float s  = sc[p][nl];
        const float zz = zscl[p][nl];
        v16h b;
        #pragma unroll
        for (int j = 0; j < 16; ++j) {
            const int kk = hi * 16 + j;
            const int q  = (qs[cb][kk * QROW + c] >> sh) & 15;
            b[j] = (_Float16)__builtin_fmaf((float)q, s, -zz);
        }

        // ---- A fragments: 16x32 f16 (two b128 LDS reads each) ----
        union { v16h v; v8h h[2]; } a0, a1;
        const _Float16* xr0 = &xs[cb][ln        * XROW + hi * 16];
        const _Float16* xr1 = &xs[cb][(ln + 16) * XROW + hi * 16];
        a0.h[0] = *(const v8h*)(xr0);
        a0.h[1] = *(const v8h*)(xr0 + 8);
        a1.h[0] = *(const v8h*)(xr1);
        a1.h[1] = *(const v8h*)(xr1 + 8);

        acc0 = __builtin_amdgcn_wmma_f32_16x16x32_f16(false, a0.v, false, b,
                                                      (short)0, acc0, false, false);
        acc1 = __builtin_amdgcn_wmma_f32_16x16x32_f16(false, a1.v, false, b,
                                                      (short)0, acc1, false, false);

        __syncthreads();                              // done reading cb before t+3 overwrites it
        cb = (cb == NBUF - 1) ? 0 : cb + 1;
    }

    // ---- epilogue: bias + store (C/D layout: elem r -> M = r + hi*8) ----
    const int   n  = n0 + wave * 16 + ln;
    const float bi = bias[n];
    #pragma unroll
    for (int r = 0; r < 8; ++r) {
        const int m = r + hi * 8;
        out[(size_t)m        * NDIM + n] = acc0[r] + bi;
        out[(size_t)(m + 16) * NDIM + n] = acc1[r] + bi;
    }
}

extern "C" void kernel_launch(void* const* d_in, const int* in_sizes, int n_in,
                              void* d_out, int out_size, void* d_ws, size_t ws_size,
                              hipStream_t stream) {
    (void)in_sizes; (void)n_in; (void)ws_size; (void)out_size;
    const float* xp   = (const float*)d_in[0];
    const int*   qw   = (const int*)  d_in[1];
    const float* scl  = (const float*)d_in[2];
    const int*   qz   = (const int*)  d_in[3];
    const float* bia  = (const float*)d_in[4];
    float*       outp = (float*)d_out;
    _Float16*    xw   = (_Float16*)d_ws;   // needs BATCH*KDIM*2 = 256KB of workspace
    // d_in[5] = blocksize (128), d_in[6] = n_dim (32000): compile-time constants here.

    // pre-pass: last-token activations f32 -> f16 (32 x 4096)
    hipLaunchKernelGGL(xcast_f16, dim3((BATCH * KDIM / 4) / 256), dim3(256), 0, stream, xp, xw);
    // main GEMM: 250 blocks x 8 wave32 waves, exact cover of 32000 columns
    hipLaunchKernelGGL(int4_lmhead_wmma, dim3(NDIM / NTILE), dim3(256), 0, stream,
                       xw, qw, scl, qz, bia, outp);
}
